// AttentionMemory_19662360281799
// MI455X (gfx1250) — matrix operand order repaired and checked by reading, not compiled, not run on hardware
//
#include <hip/hip_runtime.h>

// ---------------------------------------------------------------------------
// MI455X (gfx1250) implementation.
//   B=8, T=2048, D=2048, M=1024, R=B*T=16384.
//   Pipeline: LN(affine)->bf16 x ; W1/W2/enc ->bf16 ; k=enc*W2^T+b2 (bf16) ;
//             q=x*W1^T+b1 (bf16) ; attn=q*k^T * 1/sqrt(D) (f32) ;
//             LN(attn) ; out=(attn+mem)*0.5.
//   GEMMs: NT layout -> v_wmma_f32_16x16x32_bf16, double-buffered LDS tiles
//   staged via CDNA5 async global->LDS DMA (ASYNCcnt), 1 barrier per K-step.
// ---------------------------------------------------------------------------

typedef __bf16 v16bf __attribute__((ext_vector_type(16)));
typedef __bf16 v8bf  __attribute__((ext_vector_type(8)));
typedef __bf16 v4bf  __attribute__((ext_vector_type(4)));
typedef float  v8f   __attribute__((ext_vector_type(8)));
typedef float  v4f   __attribute__((ext_vector_type(4)));

#define LNEPS 1e-5f

// CDNA5 async global->LDS copy (GV mode): LDS[vdst] = MEM[vaddr], 16 bytes.
// Tracked by ASYNCcnt (CDNA5 ISA).
__device__ __forceinline__ void amk_async_ld_b128(unsigned lds_addr,
                                                  const void* gaddr)
{
    asm volatile("global_load_async_to_lds_b128 %0, %1, off"
                 :: "v"(lds_addr), "v"(gaddr) : "memory");
}
__device__ __forceinline__ void amk_wait_async0()
{
    asm volatile("s_wait_asynccnt 0x0" ::: "memory");
}

// ---------------------------------------------------------------------------
// Kernel A: per-row LayerNorm (affine) over D=2048, output bf16.
// ---------------------------------------------------------------------------
__global__ __launch_bounds__(256)
void amk_ln_affine_bf16(const float* __restrict__ in,
                        const float* __restrict__ gamma,
                        const float* __restrict__ beta,
                        __bf16* __restrict__ out)
{
    const int D   = 2048;
    const int row = blockIdx.x;
    const int tid = threadIdx.x;
    const float* r = in + (size_t)row * D;

    v4f a = *(const v4f*)(r + tid * 4);
    v4f b = *(const v4f*)(r + 1024 + tid * 4);

    float s = 0.f, s2 = 0.f;
#pragma unroll
    for (int i = 0; i < 4; ++i) {
        s  += a[i] + b[i];
        s2 += a[i] * a[i] + b[i] * b[i];
    }

    __shared__ float rs[256], rq[256];
    rs[tid] = s; rq[tid] = s2;
    __syncthreads();
    for (int o = 128; o > 0; o >>= 1) {
        if (tid < o) { rs[tid] += rs[tid + o]; rq[tid] += rq[tid + o]; }
        __syncthreads();
    }
    const float mean = rs[0] * (1.0f / D);
    const float var  = rq[0] * (1.0f / D) - mean * mean;
    const float rstd = rsqrtf(var + LNEPS);

    v4f g1 = *(const v4f*)(gamma + tid * 4);
    v4f g2 = *(const v4f*)(gamma + 1024 + tid * 4);
    v4f e1 = *(const v4f*)(beta + tid * 4);
    v4f e2 = *(const v4f*)(beta + 1024 + tid * 4);

    v4bf o1, o2;
#pragma unroll
    for (int i = 0; i < 4; ++i) {
        o1[i] = (__bf16)((a[i] - mean) * rstd * g1[i] + e1[i]);
        o2[i] = (__bf16)((b[i] - mean) * rstd * g2[i] + e2[i]);
    }
    __bf16* w = out + (size_t)row * D;
    *(v4bf*)(w + tid * 4)        = o1;
    *(v4bf*)(w + 1024 + tid * 4) = o2;
}

// ---------------------------------------------------------------------------
// Kernel B: f32 -> bf16 converter, 4 elements/thread. n must be mult of 1024.
// ---------------------------------------------------------------------------
__global__ __launch_bounds__(256)
void amk_cvt_bf16(const float* __restrict__ in, __bf16* __restrict__ out)
{
    const long i = ((long)blockIdx.x * 256 + threadIdx.x) * 4;
    v4f a = *(const v4f*)(in + i);
    v4bf o;
#pragma unroll
    for (int j = 0; j < 4; ++j) o[j] = (__bf16)a[j];
    *(v4bf*)(out + i) = o;
}

// ---------------------------------------------------------------------------
// Kernel C: bf16 NT GEMM.  C[m,n] = alpha * sum_k A[m,k]*Bm[n,k] (+ bias[n])
// Block = 256 threads (8 waves); block tile 128x128; wave tile 64x32
// (4 M-frags x 2 N-frags), K-step 32 via v_wmma_f32_16x16x32_bf16.
// Double-buffered LDS tiles (row stride 40 halves = 80B, 16B aligned),
// staged by global_load_async_to_lds_b128; s_wait_asynccnt 0 + one barrier
// per K-step.  M,N mult of 128; K mult of 32 -> no bounds checks.
// ---------------------------------------------------------------------------
template<bool BF16OUT, bool HASBIAS>
__global__ __launch_bounds__(256)
void amk_gemm_bf16_nt(const __bf16* __restrict__ A, int lda,
                      const __bf16* __restrict__ Bm, int ldb,
                      float* __restrict__ Cf, __bf16* __restrict__ Cb, int ldc,
                      int K, const float* __restrict__ bias, float alpha)
{
    __shared__ __align__(16) __bf16 As[2][128 * 40];
    __shared__ __align__(16) __bf16 Bs[2][128 * 40];
    const unsigned BUFB = 128 * 40 * 2;   // bytes per buffer

    const int tid  = threadIdx.x;
    const int lane = tid & 31;
    const int wave = tid >> 5;
    const int wm   = wave >> 2;     // 0..1  -> 64-row strip
    const int wn   = wave & 3;      // 0..3  -> 32-col strip

    const long rowM0 = (long)blockIdx.y * 128;
    const long rowN0 = (long)blockIdx.x * 128;

    // staging map: 256 threads cover 128 rows x 32 halves (2x 16B each matrix)
    const int ldRow = tid >> 1;
    const int ldCol = (tid & 1) << 4;   // 0 or 16 halves

    const __bf16* aptr = A  + (rowM0 + ldRow) * (long)lda + ldCol;
    const __bf16* bptr = Bm + (rowN0 + ldRow) * (long)ldb + ldCol;

    // LDS byte addresses for this thread's staging slots (flat addr low 32
    // bits == LDS offset, ISA 10.2 aperture rule).
    const unsigned stByte = (unsigned)(ldRow * 40 + ldCol) * 2u;
    const unsigned asA = (unsigned)(unsigned long long)(&As[0][0]) + stByte;
    const unsigned asB = (unsigned)(unsigned long long)(&Bs[0][0]) + stByte;

    v8f acc[4][2] = {};

    const int rsel   = lane & 15;
    const int hiHalf = lane >> 4;       // lane group 0/1
    const int aOff   = hiHalf * 8;      // A frag: K {0-7,16-23} vs {8-15,24-31}
    const int bOff   = hiHalf * 8;      // B frag: K {0-7,16-23} vs {8-15,24-31}

    // ---- prologue: stage tile 0 into buffer 0 (async DMA) ----
    amk_async_ld_b128(asA,      aptr);
    amk_async_ld_b128(asA + 16, aptr + 8);
    amk_async_ld_b128(asB,      bptr);
    amk_async_ld_b128(asB + 16, bptr + 8);
    aptr += 32; bptr += 32;
    amk_wait_async0();
    __syncthreads();

    const int niter = K >> 5;
    for (int i = 0; i < niter; ++i) {
        const int  cur  = i & 1;
        const bool more = (i + 1) < niter;

        // fire-and-forget prefetch of next K-slice straight into LDS
        if (more) {
            const unsigned bo = (unsigned)(cur ^ 1) * BUFB;
            amk_async_ld_b128(asA + bo,      aptr);
            amk_async_ld_b128(asA + bo + 16, aptr + 8);
            amk_async_ld_b128(asB + bo,      bptr);
            amk_async_ld_b128(asB + bo + 16, bptr + 8);
            aptr += 32; bptr += 32;
        }

        v16bf af[4], bfv[2];
#pragma unroll
        for (int fm = 0; fm < 4; ++fm) {
            const __bf16* p = &As[cur][(wm * 64 + fm * 16 + rsel) * 40 + aOff];
            v8bf lo = *(const v8bf*)(p);
            v8bf hi = *(const v8bf*)(p + 16);
            af[fm] = __builtin_shufflevector(lo, hi,
                0,1,2,3,4,5,6,7,8,9,10,11,12,13,14,15);
        }
#pragma unroll
        for (int fn = 0; fn < 2; ++fn) {
            const __bf16* p = &Bs[cur][(wn * 32 + fn * 16 + rsel) * 40 + bOff];
            v8bf lo = *(const v8bf*)(p);
            v8bf hi = *(const v8bf*)(p + 16);
            bfv[fn] = __builtin_shufflevector(lo, hi,
                0,1,2,3,4,5,6,7,8,9,10,11,12,13,14,15);
        }
#pragma unroll
        for (int fm = 0; fm < 4; ++fm)
#pragma unroll
            for (int fn = 0; fn < 2; ++fn)
                acc[fm][fn] = __builtin_amdgcn_wmma_f32_16x16x32_bf16(
                    false, af[fm], false, bfv[fn],
                    (short)0, acc[fm][fn], false, false);

        // own async writes into buf[nxt] must be complete before signaling;
        // barrier then publishes them to all waves for iteration i+1.
        amk_wait_async0();
        __syncthreads();
    }

    // Epilogue.  C/D frag layout: lane<16 -> rows 0-7 in vgpr j, lane>=16 -> 8-15.
#pragma unroll
    for (int fm = 0; fm < 4; ++fm) {
        const long m0 = rowM0 + wm * 64 + fm * 16 + hiHalf * 8;
#pragma unroll
        for (int fn = 0; fn < 2; ++fn) {
            const long n = rowN0 + wn * 32 + fn * 16 + rsel;
            float bv = 0.0f;
            if constexpr (HASBIAS) bv = bias[n];
            if constexpr (BF16OUT) {
                __bf16* cp = Cb + m0 * (long)ldc + n;
#pragma unroll
                for (int j = 0; j < 8; ++j) {
                    *cp = (__bf16)(acc[fm][fn][j] * alpha + bv);
                    cp += ldc;
                }
            } else {
                float* cp = Cf + m0 * (long)ldc + n;
#pragma unroll
                for (int j = 0; j < 8; ++j) {
                    *cp = acc[fm][fn][j] * alpha + bv;
                    cp += ldc;
                }
            }
        }
    }
}

// ---------------------------------------------------------------------------
// Kernel D: per-row param-free LayerNorm over M=1024 + residual*0.5 -> out.
// ---------------------------------------------------------------------------
__global__ __launch_bounds__(256)
void amk_ln_res(const float* __restrict__ attn,
                const float* __restrict__ mem,
                float* __restrict__ out)
{
    const int Md  = 1024;
    const int row = blockIdx.x;
    const int tid = threadIdx.x;
    const float* r = attn + (size_t)row * Md;

    v4f a = *(const v4f*)(r + tid * 4);
    float s = 0.f, s2 = 0.f;
#pragma unroll
    for (int i = 0; i < 4; ++i) { s += a[i]; s2 += a[i] * a[i]; }

    __shared__ float rs[256], rq[256];
    rs[tid] = s; rq[tid] = s2;
    __syncthreads();
    for (int o = 128; o > 0; o >>= 1) {
        if (tid < o) { rs[tid] += rs[tid + o]; rq[tid] += rq[tid + o]; }
        __syncthreads();
    }
    const float mean = rs[0] * (1.0f / Md);
    const float var  = rq[0] * (1.0f / Md) - mean * mean;
    const float rstd = rsqrtf(var + LNEPS);

    v4f mv = *(const v4f*)(mem + (size_t)row * Md + tid * 4);
    v4f o;
#pragma unroll
    for (int i = 0; i < 4; ++i)
        o[i] = ((a[i] - mean) * rstd + mv[i]) * 0.5f;
    *(v4f*)(out + (size_t)row * Md + tid * 4) = o;
}

// ---------------------------------------------------------------------------
extern "C" void kernel_launch(void* const* d_in, const int* in_sizes, int n_in,
                              void* d_out, int out_size, void* d_ws, size_t ws_size,
                              hipStream_t stream)
{
    const float* dec   = (const float*)d_in[0];
    const float* enc   = (const float*)d_in[1];
    const float* mem   = (const float*)d_in[2];
    const float* gamma = (const float*)d_in[3];
    const float* beta  = (const float*)d_in[4];
    const float* W1    = (const float*)d_in[5];
    const float* b1    = (const float*)d_in[6];
    const float* W2    = (const float*)d_in[7];
    const float* b2    = (const float*)d_in[8];
    float* out = (float*)d_out;

    const int D = 2048, Md = 1024, R = 8 * 2048;   // R = B*T = 16384

    // Workspace layout (region0 reused: x_bf16 then attn_f32, both 64 MiB).
    char* base = (char*)d_ws;
    size_t off = 0;
    auto wsalloc = [&](size_t bytes) -> void* {
        void* p = base + off;
        off += (bytes + 255) & ~(size_t)255;
        return p;
    };
    __bf16* x_bf   = (__bf16*)wsalloc((size_t)R * D * sizeof(__bf16));  // 64 MiB
    float*  attn_f = (float*)x_bf;                  // alias (same 64 MiB size)
    __bf16* q_bf   = (__bf16*)wsalloc((size_t)R * D * sizeof(__bf16));  // 64 MiB
    __bf16* w1_bf  = (__bf16*)wsalloc((size_t)D * D * sizeof(__bf16));
    __bf16* w2_bf  = (__bf16*)wsalloc((size_t)D * D * sizeof(__bf16));
    __bf16* enc_bf = (__bf16*)wsalloc((size_t)Md * D * sizeof(__bf16));
    __bf16* k_bf   = (__bf16*)wsalloc((size_t)Md * D * sizeof(__bf16));
    (void)ws_size; (void)in_sizes; (void)n_in; (void)out_size;

    const float invTemp = 1.0f / sqrtf((float)D);

    // 1) x = LN(dec)*gamma+beta -> bf16
    amk_ln_affine_bf16<<<R, 256, 0, stream>>>(dec, gamma, beta, x_bf);

    // 2) bf16 conversions of weights / enc
    amk_cvt_bf16<<<(D * D) / 1024, 256, 0, stream>>>(W1, w1_bf);
    amk_cvt_bf16<<<(D * D) / 1024, 256, 0, stream>>>(W2, w2_bf);
    amk_cvt_bf16<<<(Md * D) / 1024, 256, 0, stream>>>(enc, enc_bf);

    // 3) k = enc * W2^T + b2   [Md x D], bf16 out
    amk_gemm_bf16_nt<true, true><<<dim3(D / 128, Md / 128), 256, 0, stream>>>(
        enc_bf, D, w2_bf, D, nullptr, k_bf, D, D, b2, 1.0f);

    // 4) q = x * W1^T + b1     [R x D], bf16 out
    amk_gemm_bf16_nt<true, true><<<dim3(D / 128, R / 128), 256, 0, stream>>>(
        x_bf, D, w1_bf, D, nullptr, q_bf, D, D, b1, 1.0f);

    // 5) attn = q * k^T / sqrt(D)   [R x Md], f32 out (reuses x region)
    amk_gemm_bf16_nt<false, false><<<dim3(Md / 128, R / 128), 256, 0, stream>>>(
        q_bf, D, k_bf, D, attn_f, (__bf16*)nullptr, Md, D, nullptr, invTemp);

    // 6) out = (LN(attn) + mem) * 0.5
    amk_ln_res<<<R, 256, 0, stream>>>(attn_f, mem, out);
}
